// HierarchicalTreeGating_46523085750366
// MI455X (gfx1250) — matrix operand order repaired and checked by reading, not compiled, not run on hardware
//
#include <hip/hip_runtime.h>
#include <hip/hip_bf16.h>
#include <math.h>

#define FEAT    1024
#define HID     64
#define NNODES  7
#define NSCALES 5

typedef __attribute__((ext_vector_type(16))) __bf16 v16bf;
typedef __attribute__((ext_vector_type(8)))  __bf16 v8bf;
typedef __attribute__((ext_vector_type(4)))  __bf16 v4bf;
typedef __attribute__((ext_vector_type(8)))  float  v8f;
typedef __attribute__((ext_vector_type(4)))  float  f4;

#define WMMA_BF16(A, B, C) \
    __builtin_amdgcn_wmma_f32_16x16x32_bf16(false, (A), false, (B), (short)0, (C), false, false)

__device__ __forceinline__ float gelu_exact(float x) {
    return 0.5f * x * (1.0f + erff(x * 0.70710678118654752440f));
}

// butterfly sum across each 16-lane half of the wave32
__device__ __forceinline__ float redux16(float v) {
    v += __shfl_xor(v, 1, 32);
    v += __shfl_xor(v, 2, 32);
    v += __shfl_xor(v, 4, 32);
    v += __shfl_xor(v, 8, 32);
    return v;
}

// ---- B-matrix frag (32x16 KxN, contiguous k, same as A): lane holds column n0, 16
// consecutive K values starting at half*16.
__device__ __forceinline__ v16bf load_w_frag(const float* __restrict__ p) {
    const f4* q = (const f4*)p;
    f4 x0 = q[0], x1 = q[1], x2 = q[2], x3 = q[3];
    v16bf v;
    v[0]  = (__bf16)x0.x; v[1]  = (__bf16)x0.y; v[2]  = (__bf16)x0.z; v[3]  = (__bf16)x0.w;
    v[4]  = (__bf16)x1.x; v[5]  = (__bf16)x1.y; v[6]  = (__bf16)x1.z; v[7]  = (__bf16)x1.w;
    v[8]  = (__bf16)x2.x; v[9]  = (__bf16)x2.y; v[10] = (__bf16)x2.z; v[11] = (__bf16)x2.w;
    v[12] = (__bf16)x3.x; v[13] = (__bf16)x3.y; v[14] = (__bf16)x3.z; v[15] = (__bf16)x3.w;
    return v;
}
__device__ __forceinline__ v16bf load_w_frag(const __bf16* __restrict__ p) {
    const v8bf* q = (const v8bf*)p;
    v8bf a = q[0], b = q[1];               // two b128 loads, no cvt
    return __builtin_shufflevector(a, b, 0,1,2,3,4,5,6,7,8,9,10,11,12,13,14,15);
}

// ---- A-matrix frag (16x32 MxK, contiguous k, same as B): lane holds row m, 16
// consecutive K values starting at k0 = kc + half*16.
__device__ __forceinline__ v16bf load_a_frag(const float* __restrict__ p) {
    const f4* q = (const f4*)p;
    f4 x0 = q[0], x1 = q[1];      // k0 .. k0+7
    f4 x2 = q[2], x3 = q[3];      // k0+8 .. k0+15
    v16bf v;
    v[0]  = (__bf16)x0.x; v[1]  = (__bf16)x0.y; v[2]  = (__bf16)x0.z; v[3]  = (__bf16)x0.w;
    v[4]  = (__bf16)x1.x; v[5]  = (__bf16)x1.y; v[6]  = (__bf16)x1.z; v[7]  = (__bf16)x1.w;
    v[8]  = (__bf16)x2.x; v[9]  = (__bf16)x2.y; v[10] = (__bf16)x2.z; v[11] = (__bf16)x2.w;
    v[12] = (__bf16)x3.x; v[13] = (__bf16)x3.y; v[14] = (__bf16)x3.z; v[15] = (__bf16)x3.w;
    return v;
}
__device__ __forceinline__ v16bf load_a_frag(const __bf16* __restrict__ p) {
    v8bf a = *(const v8bf*)p;            // k0 .. k0+7
    v8bf b = *(const v8bf*)(p + 8);      // k0+8 .. k0+15
    return __builtin_shufflevector(a, b, 0,1,2,3,4,5,6,7,8,9,10,11,12,13,14,15);
}

// f32 -> bf16 pre-conversion (4 elements / thread)
__global__ __launch_bounds__(256) void cvt_bf16_kernel(
    const float* __restrict__ src, __bf16* __restrict__ dst, int n4)
{
    const int i = blockIdx.x * 256 + threadIdx.x;
    if (i < n4) {
        f4 x = ((const f4*)src)[i];
        v4bf y;
        y[0] = (__bf16)x.x; y[1] = (__bf16)x.y; y[2] = (__bf16)x.z; y[3] = (__bf16)x.w;
        ((v4bf*)dst)[i] = y;
    }
}

// One wave (32 threads) handles 16 batch rows. AT/WT = float (cvt on the fly)
// or __bf16 (pre-converted operands in workspace).
template <typename AT, typename WT>
__global__ __launch_bounds__(32) void gates_kernel(
    const AT* __restrict__ feat,
    const WT* __restrict__ tw1, const float* __restrict__ tb1,
    const float* __restrict__ tg,  const float* __restrict__ tbn,
    const float* __restrict__ tw2, const float* __restrict__ tb2,
    const float* __restrict__ lw1, const float* __restrict__ lb1,
    const float* __restrict__ lw2, const float* __restrict__ lb2,
    const WT* __restrict__ dw1, const float* __restrict__ db1,
    const float* __restrict__ dw2, const float* __restrict__ db2,
    const float* __restrict__ cw,
    float* __restrict__ gates_out)
{
    const int lane  = threadIdx.x & 31;
    const int n0    = lane & 15;        // column / A-row within tile
    const int half  = lane >> 4;        // lane half selects K sub-range / M+8
    const int mbase = blockIdx.x * 16;

    const AT* __restrict__ arow = feat + (size_t)(mbase + n0) * FEAT + half * 16;

    __shared__ float nodeP[16][NNODES][2];  // per-row node child probabilities
    __shared__ float dlog[16][NSCALES];     // per-row direct-path logits (pre-bias)

    // ---------- Phase 1: tree expert GEMM, one node (64 cols = 4 N-tiles) at a time
    for (int n = 0; n < NNODES; ++n) {
        v8f cc[4] = {};
        const WT* __restrict__ wb = tw1 + ((size_t)n * HID + n0) * FEAT + half * 16;
        for (int kc = 0; kc < FEAT; kc += 32) {
            v16bf a = load_a_frag(arow + kc);
#pragma unroll
            for (int t = 0; t < 4; ++t)
                cc[t] = WMMA_BF16(a, load_w_frag(wb + (size_t)t * 16 * FEAT + kc), cc[t]);
        }
        // epilogue: + bias, LayerNorm over 64, gelu, 64->2, softmax(T=0.5)
        float hv[4][8];
        float gam[4], bet[4], w2a[4], w2b[4];
#pragma unroll
        for (int t = 0; t < 4; ++t) {
            const int hid = n * HID + t * 16 + n0;
            const float bias = tb1[hid];
            gam[t] = tg[hid];
            bet[t] = tbn[hid];
            w2a[t] = tw2[(size_t)n * 2 * HID +       t * 16 + n0];
            w2b[t] = tw2[(size_t)n * 2 * HID + HID + t * 16 + n0];
#pragma unroll
            for (int vr = 0; vr < 8; ++vr) hv[t][vr] = cc[t][vr] + bias;
        }
        const float bl0 = tb2[n * 2 + 0];
        const float bl1 = tb2[n * 2 + 1];
#pragma unroll
        for (int vr = 0; vr < 8; ++vr) {
            float s = 0.f, q = 0.f;
#pragma unroll
            for (int t = 0; t < 4; ++t) { s += hv[t][vr]; q += hv[t][vr] * hv[t][vr]; }
            s = redux16(s);
            q = redux16(q);
            const float mean = s * (1.0f / 64.0f);
            const float var  = q * (1.0f / 64.0f) - mean * mean;
            const float rstd = rsqrtf(var + 1e-5f);
            float p0 = 0.f, p1 = 0.f;
#pragma unroll
            for (int t = 0; t < 4; ++t) {
                float x = (hv[t][vr] - mean) * rstd * gam[t] + bet[t];
                x = gelu_exact(x);
                p0 += x * w2a[t];
                p1 += x * w2b[t];
            }
            p0 = redux16(p0) + bl0;
            p1 = redux16(p1) + bl1;
            const float pL = 1.0f / (1.0f + expf(2.0f * (p1 - p0)));  // softmax at T=0.5
            if (n0 == 0) {
                nodeP[half * 8 + vr][n][0] = pL;
                nodeP[half * 8 + vr][n][1] = 1.0f - pL;
            }
        }
    }

    // ---------- Phase 2: direct path GEMM (256 cols = 16 N-tiles), K-outer
    {
        v8f acc[16] = {};
        const WT* __restrict__ wb = dw1 + (size_t)n0 * FEAT + half * 16;
        for (int kc = 0; kc < FEAT; kc += 32) {
            v16bf a = load_a_frag(arow + kc);
#pragma unroll
            for (int j = 0; j < 16; ++j)
                acc[j] = WMMA_BF16(a, load_w_frag(wb + (size_t)j * 16 * FEAT + kc), acc[j]);
        }
        float pd[NSCALES][8] = {};
#pragma unroll
        for (int j = 0; j < 16; ++j) {
            const int col = j * 16 + n0;
            const float bias = db1[col];
            const float w0 = dw2[0 * 256 + col];
            const float w1 = dw2[1 * 256 + col];
            const float w2 = dw2[2 * 256 + col];
            const float w3 = dw2[3 * 256 + col];
            const float w4 = dw2[4 * 256 + col];
#pragma unroll
            for (int vr = 0; vr < 8; ++vr) {
                const float g = gelu_exact(acc[j][vr] + bias);
                pd[0][vr] += g * w0;
                pd[1][vr] += g * w1;
                pd[2][vr] += g * w2;
                pd[3][vr] += g * w3;
                pd[4][vr] += g * w4;
            }
        }
#pragma unroll
        for (int s = 0; s < NSCALES; ++s)
#pragma unroll
            for (int vr = 0; vr < 8; ++vr) {
                const float v = redux16(pd[s][vr]);
                if (n0 == 0) dlog[half * 8 + vr][s] = v;
            }
    }
    __syncthreads();

    // ---------- Phase 3: per-row scalar tail (one thread per row)
    if (threadIdx.x < 16) {
        const int r = threadIdx.x;
        float P[NNODES][2];
#pragma unroll
        for (int nn = 0; nn < NNODES; ++nn) {
            P[nn][0] = nodeP[r][nn][0];
            P[nn][1] = nodeP[r][nn][1];
        }
        float lp[8];
#pragma unroll
        for (int l = 0; l < 8; ++l) {
            const int b2 = (l >> 2) & 1, b1 = (l >> 1) & 1, b0 = l & 1;
            lp[l] = P[0][b2] * P[1 + b2][b1] * P[3 + (l >> 1)][b0];
        }
        float t10[10];
#pragma unroll
        for (int i = 0; i < 10; ++i) {
            float a = lb1[i];
#pragma unroll
            for (int l = 0; l < 8; ++l) a += lp[l] * lw1[i * 8 + l];
            t10[i] = gelu_exact(a);
        }
        float tl[NSCALES];
#pragma unroll
        for (int s = 0; s < NSCALES; ++s) {
            float v = lb2[s];
#pragma unroll
            for (int i = 0; i < 10; ++i) v += t10[i] * lw2[s * 10 + i];
            tl[s] = v;
        }
        // softmax(tree)
        float mx = tl[0];
#pragma unroll
        for (int s = 1; s < NSCALES; ++s) mx = fmaxf(mx, tl[s]);
        float tgate[NSCALES], se = 0.f;
#pragma unroll
        for (int s = 0; s < NSCALES; ++s) { tgate[s] = expf(tl[s] - mx); se += tgate[s]; }
        const float ise = 1.0f / se;
        // softmax(direct)
        float dl[NSCALES];
#pragma unroll
        for (int s = 0; s < NSCALES; ++s) dl[s] = dlog[r][s] + db2[s];
        float mxd = dl[0];
#pragma unroll
        for (int s = 1; s < NSCALES; ++s) mxd = fmaxf(mxd, dl[s]);
        float dgate[NSCALES], sd = 0.f;
#pragma unroll
        for (int s = 0; s < NSCALES; ++s) { dgate[s] = expf(dl[s] - mxd); sd += dgate[s]; }
        const float isd = 1.0f / sd;

        const float w = 1.0f / (1.0f + expf(-cw[0]));
        float g[NSCALES], gsum = 0.f;
#pragma unroll
        for (int s = 0; s < NSCALES; ++s) {
            g[s] = w * tgate[s] * ise + (1.0f - w) * dgate[s] * isd;
            gsum += g[s];
        }
        const float inv = 1.0f / gsum;
        float* __restrict__ go = gates_out + (size_t)(mbase + r) * NSCALES;
#pragma unroll
        for (int s = 0; s < NSCALES; ++s) go[s] = g[s] * inv;
    }
}

// Streaming combine: out[b,:] = sum_s gates[b,s] * logits[s,b,:]
// Non-temporal b128 traffic; gates are block-uniform scalar loads.
__global__ __launch_bounds__(256) void combine_kernel(
    const float* __restrict__ logits, const float* __restrict__ gates,
    float* __restrict__ out, int blocksPerRow, int f4PerRow, size_t planeF4)
{
    const int b = blockIdx.x / blocksPerRow;
    const int c = (blockIdx.x % blocksPerRow) * 256 + threadIdx.x;  // float4 idx in row
    const float* __restrict__ g = gates + (size_t)b * NSCALES;
    const float g0 = g[0], g1 = g[1], g2 = g[2], g3 = g[3], g4 = g[4];
    const size_t base = (size_t)b * f4PerRow + c;
    const f4* __restrict__ L = (const f4*)logits;
    f4 v0 = __builtin_nontemporal_load(L + base);
    f4 v1 = __builtin_nontemporal_load(L + base + planeF4);
    f4 v2 = __builtin_nontemporal_load(L + base + 2 * planeF4);
    f4 v3 = __builtin_nontemporal_load(L + base + 3 * planeF4);
    f4 v4 = __builtin_nontemporal_load(L + base + 4 * planeF4);
    f4 r = g0 * v0 + g1 * v1 + g2 * v2 + g3 * v3 + g4 * v4;
    __builtin_nontemporal_store(r, ((f4*)out) + base);
}

extern "C" void kernel_launch(void* const* d_in, const int* in_sizes, int n_in,
                              void* d_out, int out_size, void* d_ws, size_t ws_size,
                              hipStream_t stream) {
    const float* feat   = (const float*)d_in[0];
    const float* logits = (const float*)d_in[1];
    const float* tw1    = (const float*)d_in[2];
    const float* tb1    = (const float*)d_in[3];
    const float* tg     = (const float*)d_in[4];
    const float* tbn    = (const float*)d_in[5];
    const float* tw2    = (const float*)d_in[6];
    const float* tb2    = (const float*)d_in[7];
    const float* lw1    = (const float*)d_in[8];
    const float* lb1    = (const float*)d_in[9];
    const float* lw2    = (const float*)d_in[10];
    const float* lb2    = (const float*)d_in[11];
    const float* dw1    = (const float*)d_in[12];
    const float* db1    = (const float*)d_in[13];
    const float* dw2    = (const float*)d_in[14];
    const float* db2    = (const float*)d_in[15];
    const float* cw     = (const float*)d_in[16];

    const int B = in_sizes[0] / FEAT;                       // 8192
    const long long tot = (long long)in_sizes[1];           // 5*B*D
    const int D = (int)(tot / ((long long)NSCALES * B));    // 4096

    float* out_comb  = (float*)d_out;
    float* out_gates = (float*)d_out + (size_t)B * D;

    const size_t nFeat = (size_t)in_sizes[0];    // B*1024
    const size_t nTw1  = (size_t)in_sizes[2];    // 7*64*1024
    const size_t nDw1  = (size_t)in_sizes[12];   // 256*1024
    const size_t need  = (nFeat + nTw1 + nDw1) * sizeof(__bf16);

    if (ws_size >= need) {
        // Pre-convert GEMM operands to bf16: halves operand bandwidth and
        // removes all cvt work from the wmma inner loops.
        __bf16* featb = (__bf16*)d_ws;
        __bf16* tw1b  = featb + nFeat;
        __bf16* dw1b  = tw1b + nTw1;
        const int nf4 = (int)(nFeat / 4), nt4 = (int)(nTw1 / 4), nd4 = (int)(nDw1 / 4);
        cvt_bf16_kernel<<<dim3((nf4 + 255) / 256), dim3(256), 0, stream>>>(feat, featb, nf4);
        cvt_bf16_kernel<<<dim3((nt4 + 255) / 256), dim3(256), 0, stream>>>(tw1, tw1b, nt4);
        cvt_bf16_kernel<<<dim3((nd4 + 255) / 256), dim3(256), 0, stream>>>(dw1, dw1b, nd4);
        gates_kernel<__bf16, __bf16><<<dim3(B / 16), dim3(32), 0, stream>>>(
            featb, tw1b, tb1, tg, tbn, tw2, tb2, lw1, lb1, lw2, lb2,
            dw1b, db1, dw2, db2, cw, out_gates);
    } else {
        gates_kernel<float, float><<<dim3(B / 16), dim3(32), 0, stream>>>(
            feat, tw1, tb1, tg, tbn, tw2, tb2, lw1, lb1, lw2, lb2,
            dw1, db1, dw2, db2, cw, out_gates);
    }

    const int f4PerRow     = D / 4;            // 1024
    const int blocksPerRow = f4PerRow / 256;   // 4
    const size_t planeF4   = (size_t)B * f4PerRow;
    combine_kernel<<<dim3(B * blocksPerRow), dim3(256), 0, stream>>>(
        logits, out_gates, out_comb, blocksPerRow, f4PerRow, planeF4);
}
